// QuantizedExpertFusedCpuLinear_48954037240358
// MI455X (gfx1250) — compile-verified, matrix operand order checked by reading, not run on hardware
//
#include <hip/hip_runtime.h>

typedef __attribute__((ext_vector_type(16))) _Float16 v16h;
typedef __attribute__((ext_vector_type(8)))  _Float16 v8h;
typedef __attribute__((ext_vector_type(4)))  _Float16 v4h;
typedef __attribute__((ext_vector_type(8)))  float    v8f;
typedef __attribute__((ext_vector_type(4)))  float    f32x4;
typedef __attribute__((ext_vector_type(4)))  int      i32x4;

#define NE    8
#define NT    1024
#define DIN   2048
#define DOUT  8192

#define BM    128          // token tile
#define BN    128          // output-feature tile
#define BK    32           // K tile == one v_wmma_f32_16x16x32_f16 step
#define PAD   8            // halves of padding -> 80B row stride, conflict-free b128
#define LDK   (BK + PAD)   // 40 halves per LDS row
#define NSTAGES (DIN / BK) // 64

__global__ __launch_bounds__(256, 2)
void qexpert_wmma_gemm(const float* __restrict__ A,   // [E, T, DIN] f32
                       const int*   __restrict__ Wq,  // [E, DOUT, DIN] int32 (int8-valued)
                       const float* __restrict__ Sc,  // [E]
                       float*       __restrict__ C)   // [E, T, DOUT] f32
{
    __shared__ _Float16 sA[2][BM][LDK];
    __shared__ _Float16 sB[2][BN][LDK];

    const int e    = blockIdx.z;
    const int m0   = blockIdx.y * BM;
    const int n0   = blockIdx.x * BN;
    const int tid  = threadIdx.x;
    const int lane = tid & 31;
    const int w    = tid >> 5;         // 8 waves
    const int wm   = (w & 1) * 64;     // wave M offset inside block tile
    const int wn   = (w >> 1) * 32;    // wave N offset inside block tile

    const float* Ab = A  + (size_t)e * NT   * DIN;
    const int*   Bb = Wq + (size_t)e * DOUT * DIN;

    // ---- global->register staging (b128), 4 chunks each for A and B ----
    // chunk c in [0,1024): row = c>>3, kc = (c&7)*4 ; thread owns c = tid + 256*j
    f32x4 aReg[4];
    i32x4 bReg[4];

    auto gload = [&](int ks) {
        const int kb = ks * BK;
#pragma unroll
        for (int j = 0; j < 4; ++j) {
            const int c   = tid + 256 * j;
            const int row = c >> 3;
            const int kc  = (c & 7) * 4;
            aReg[j] = *(const f32x4*)(Ab + (size_t)(m0 + row) * DIN + kb + kc);
            bReg[j] = *(const i32x4*)(Bb + (size_t)(n0 + row) * DIN + kb + kc);
        }
    };

    auto lstore = [&](int buf) {
#pragma unroll
        for (int j = 0; j < 4; ++j) {
            const int c   = tid + 256 * j;
            const int row = c >> 3;
            const int kc  = (c & 7) * 4;
            v4h ha, hb;
#pragma unroll
            for (int i = 0; i < 4; ++i) {
                ha[i] = (_Float16)aReg[j][i];
                hb[i] = (_Float16)(float)bReg[j][i];
            }
            *(v4h*)&sA[buf][row][kc] = ha;   // ds_store_b64, 8B aligned
            *(v4h*)&sB[buf][row][kc] = hb;
        }
    };

    // ---- WMMA fragment load per CDNA5 16-bit A/B layout ----
    // lane L: row = L&15 ; K chunks at (L>=16 ? 8 : 0) and +16
    const int fRow = lane & 15;
    const int kSel = (lane >> 4) * 8;

    auto loadFrag = [&](const _Float16 (*s)[LDK], int row) -> v16h {
        v8h lo = *(const v8h*)&s[row][kSel];        // ds_load_b128
        v8h hi = *(const v8h*)&s[row][kSel + 16];   // ds_load_b128
        return __builtin_shufflevector(lo, hi, 0,1,2,3,4,5,6,7,
                                               8,9,10,11,12,13,14,15);
    };

    v8f acc[4][2];
#pragma unroll
    for (int mi = 0; mi < 4; ++mi)
#pragma unroll
        for (int ni = 0; ni < 2; ++ni)
            acc[mi][ni] = (v8f){};

    gload(0);

    for (int ks = 0; ks < NSTAGES; ++ks) {
        const int buf = ks & 1;
        // convert staged regs -> f16 LDS tiles (other threads consume after sync)
        lstore(buf);
        __syncthreads();   // single barrier per stage: buffer b written at ks is
                           // only read at ks; next write of b (ks+2) is fenced by
                           // the sync inside stage ks+1.
        if (ks + 1 < NSTAGES) gload(ks + 1);   // prefetch hidden behind compute

        v16h af[4], bf[2];
#pragma unroll
        for (int mi = 0; mi < 4; ++mi)
            af[mi] = loadFrag(sA[buf], wm + mi * 16 + fRow);
#pragma unroll
        for (int ni = 0; ni < 2; ++ni)
            bf[ni] = loadFrag(sB[buf], wn + ni * 16 + fRow);

#pragma unroll
        for (int mi = 0; mi < 4; ++mi)
#pragma unroll
            for (int ni = 0; ni < 2; ++ni)
                acc[mi][ni] = __builtin_amdgcn_wmma_f32_16x16x32_f16(
                    false, af[mi], false, bf[ni],
                    (short)0, acc[mi][ni], false, false);
    }

    // ---- epilogue: apply per-expert scale, store f32 ----
    // C/D layout: VGPR r of lane L -> M = r + 8*(L>=16), N = L&15
    const float s     = Sc[e];
    const int   mLane = (lane >> 4) * 8;
    const int   nLane = lane & 15;
    float* Cb = C + (size_t)e * NT * DOUT;

#pragma unroll
    for (int mi = 0; mi < 4; ++mi) {
#pragma unroll
        for (int ni = 0; ni < 2; ++ni) {
            const int nIdx = n0 + wn + ni * 16 + nLane;
#pragma unroll
            for (int r = 0; r < 8; ++r) {
                const int mIdx = m0 + wm + mi * 16 + mLane + r;
                Cb[(size_t)mIdx * DOUT + nIdx] = acc[mi][ni][r] * s;
            }
        }
    }
}

extern "C" void kernel_launch(void* const* d_in, const int* in_sizes, int n_in,
                              void* d_out, int out_size, void* d_ws, size_t ws_size,
                              hipStream_t stream) {
    const float* input = (const float*)d_in[0];   // [E, T, DIN] f32
    const int*   wq    = (const int*)d_in[1];     // [E, DOUT, DIN] int32
    const float* scale = (const float*)d_in[2];   // [E] f32
    float*       out   = (float*)d_out;           // [E, T, DOUT] f32

    dim3 grid(DOUT / BN, NT / BM, NE);            // 64 x 8 x 8 = 4096 blocks
    qexpert_wmma_gemm<<<grid, 256, 0, stream>>>(input, wq, scale, out);
}
